// AttGCNN_69234872811972
// MI455X (gfx1250) — matrix operand order repaired and checked by reading, not compiled, not run on hardware
//
#include <hip/hip_runtime.h>
#include <hip/hip_bf16.h>
#include <math.h>

typedef __attribute__((ext_vector_type(16))) _Float16 v16h;
typedef __attribute__((ext_vector_type(8)))  _Float16 v8h;
typedef __attribute__((ext_vector_type(8)))  float    v8f;

#define TPB 256
#define WPAD 136   // padded column stride (halves): 272B = 68 dwords -> conflict-free b128 loads

// ---------- helpers ----------
__device__ __forceinline__ unsigned enc_f32(float f) {
    unsigned u = __float_as_uint(f);
    return (u & 0x80000000u) ? ~u : (u | 0x80000000u);
}
__device__ __forceinline__ float dec_f32(unsigned e) {
    // enc(x>=0) = bits|MSB -> dec: e & 0x7FFFFFFF ; enc(x<0) = ~bits -> dec: ~e
    return __uint_as_float((e & 0x80000000u) ? (e & 0x7FFFFFFFu) : ~e);
}
__device__ __forceinline__ float lrelu(float v) { return v > 0.f ? v : 0.2f * v; }

__device__ __forceinline__ void edge_endpoints(long long e, const int* __restrict__ ei,
                                               int E, int& s, int& d) {
    if (e < E) { s = ei[e]; d = ei[E + e]; }
    else       { s = (int)(e - E); d = s; }
}

// ---------- elementwise / init ----------
__global__ void k_zero_f32(float* p, long long n) {
    long long i = (long long)blockIdx.x * blockDim.x + threadIdx.x;
    if (i < n) p[i] = 0.f;
}
__global__ void k_zero_u32(unsigned* p, long long n) {
    long long i = (long long)blockIdx.x * blockDim.x + threadIdx.x;
    if (i < n) p[i] = 0u; // enc-space minimum
}
__global__ void k_cvt_f16(const float* __restrict__ src, _Float16* __restrict__ dst, long long n) {
    long long i = (long long)blockIdx.x * blockDim.x + threadIdx.x;
    if (i < n) dst[i] = (_Float16)src[i];
}

// ---------- layer-1 GEMM: h1[N,128] = x[N,128] @ W1[128,128] via WMMA ----------
// One workgroup = 16 rows of A; 8 waves each own one 16-col C tile.
// W1 staged TRANSPOSED in LDS (sWt[col][k], padded stride) so each lane's
// B fragment is two contiguous 16B ds_load_b128 reads, conflict-free.
__global__ void k_gemm1_wmma(const _Float16* __restrict__ xh,
                             const _Float16* __restrict__ Wh,
                             float* __restrict__ h1) {
    __shared__ _Float16 sWt[128 * WPAD]; // ~34 KB

    const int tid = threadIdx.x;
    // stage W1 transposed: each unit = 8 contiguous halves of a row k
    for (int u = tid; u < (128 * 128) / 8; u += TPB) {
        int k    = u >> 4;           // row of W1
        int col0 = (u & 15) << 3;    // first of 8 columns
        v8h w = *(const v8h*)(Wh + (long long)k * 128 + col0);
#pragma unroll
        for (int j = 0; j < 8; ++j) sWt[(col0 + j) * WPAD + k] = w[j];
    }
    __syncthreads();

    const int wave = tid >> 5;
    const int lane = tid & 31;
    const int m0 = blockIdx.x * 16;
    const int n0 = wave * 16;
    const int grp = lane >> 4;
    const int l15 = lane & 15;

    v8f acc;
#pragma unroll
    for (int i = 0; i < 8; ++i) acc[i] = 0.f;

    const _Float16* xrow = xh + (long long)(m0 + l15) * 128;
    const _Float16* wcol = sWt + (n0 + l15) * WPAD;

#pragma unroll
    for (int kk = 0; kk < 4; ++kk) {
        // A fragment (16-bit A 16x32 layout): lanes 0-15 hold K blocks {0..7,16..23},
        // lanes 16-31 hold {8..15,24..31} of this 32-wide K chunk.
        v8h alo = *(const v8h*)(xrow + kk * 32 + grp * 8);
        v8h ahi = *(const v8h*)(xrow + kk * 32 + 16 + grp * 8);
        v16h a = __builtin_shufflevector(alo, ahi,
                 0, 1, 2, 3, 4, 5, 6, 7, 8, 9, 10, 11, 12, 13, 14, 15);

        // B fragment: lane -> column n0+l15; lanes 0-15 hold K blocks {0..7,16..23},
        // lanes 16-31 hold {8..15,24..31} (same K order as A).
        v8h blo = *(const v8h*)(wcol + kk * 32 + grp * 8);
        v8h bhi = *(const v8h*)(wcol + kk * 32 + 16 + grp * 8);
        v16h b = __builtin_shufflevector(blo, bhi,
                 0, 1, 2, 3, 4, 5, 6, 7, 8, 9, 10, 11, 12, 13, 14, 15);

        acc = __builtin_amdgcn_wmma_f32_16x16x32_f16(
            false, a, false, b, (short)0, acc, false, false);
    }

    // C layout: VGPR r, lanes 0-15 -> M=r, lanes 16-31 -> M=r+8; N = lane&15
    const int rbase = m0 + (grp ? 8 : 0);
    const int ccol  = n0 + l15;
#pragma unroll
    for (int r = 0; r < 8; ++r)
        h1[(long long)(rbase + r) * 128 + ccol] = acc[r];
}

// ---------- attention scores, layer 1 (heads=2, C=64) ----------
__global__ void k_att1(const float* __restrict__ h1,
                       const float* __restrict__ att_src, const float* __restrict__ att_dst,
                       float* __restrict__ as1, float* __restrict__ ad1, int N) {
    int i = blockIdx.x * blockDim.x + threadIdx.x; // i = n*2 + h
    if (i >= N * 2) return;
    int n = i >> 1, h = i & 1;
    const float* row = h1 + (long long)n * 128 + h * 64;
    const float* ws = att_src + h * 64;
    const float* wd = att_dst + h * 64;
    float s = 0.f, d = 0.f;
#pragma unroll 4
    for (int c = 0; c < 64; ++c) { float v = row[c]; s += v * ws[c]; d += v * wd[c]; }
    as1[i] = s; ad1[i] = d;
}

// ---------- layer-1 edge passes ----------
__global__ void k_edge_max1(const int* __restrict__ ei, int E, long long Etot,
                            const float* __restrict__ as1, const float* __restrict__ ad1,
                            float* __restrict__ ev, unsigned* __restrict__ menc) {
    long long e = (long long)blockIdx.x * blockDim.x + threadIdx.x;
    if (e >= Etot) return;
    int s, d; edge_endpoints(e, ei, E, s, d);
#pragma unroll
    for (int h = 0; h < 2; ++h) {
        float lg = lrelu(as1[s * 2 + h] + ad1[d * 2 + h]);
        ev[e * 2 + h] = lg;
        atomicMax(&menc[d * 2 + h], enc_f32(lg));
    }
}

__global__ void k_edge_exp1(const int* __restrict__ ei, int E, long long Etot,
                            const unsigned* __restrict__ menc,
                            float* __restrict__ ev, float* __restrict__ ssum) {
    long long e = (long long)blockIdx.x * blockDim.x + threadIdx.x;
    if (e >= Etot) return;
    int s, d; edge_endpoints(e, ei, E, s, d);
#pragma unroll
    for (int h = 0; h < 2; ++h) {
        float m = dec_f32(menc[d * 2 + h]);
        float ex = __expf(ev[e * 2 + h] - m);
        ev[e * 2 + h] = ex;
        atomicAdd(&ssum[d * 2 + h], ex);
    }
}

// one thread per (edge, channel); 128 channels, head = c>>6
__global__ void k_edge_msg1(const int* __restrict__ ei, int E, long long total,
                            const float* __restrict__ h1, const float* __restrict__ ev,
                            const float* __restrict__ ssum, float* __restrict__ out1) {
    long long idx = (long long)blockIdx.x * blockDim.x + threadIdx.x;
    if (idx >= total) return;
    long long e = idx >> 7;
    int c = (int)(idx & 127);
    int h = c >> 6;
    int s, d; edge_endpoints(e, ei, E, s, d);
    float alpha = ev[e * 2 + h] / (ssum[d * 2 + h] + 1e-16f);
    atomicAdd(&out1[(long long)d * 128 + c], h1[(long long)s * 128 + c] * alpha);
}

__global__ void k_elu_bias(float* __restrict__ out1, const float* __restrict__ b1, long long n) {
    long long i = (long long)blockIdx.x * blockDim.x + threadIdx.x;
    if (i >= n) return;
    float v = out1[i] + b1[i & 127];
    out1[i] = v > 0.f ? v : (__expf(v) - 1.f);
}

// ---------- layer 2: h2 = elu(out1) @ W2[128,4]; scores (heads=1, C=4) ----------
__global__ void k_layer2_node(const float* __restrict__ x2, const float* __restrict__ W2,
                              const float* __restrict__ att_src, const float* __restrict__ att_dst,
                              float* __restrict__ h2, float* __restrict__ as2,
                              float* __restrict__ ad2, int N) {
    int n = blockIdx.x * blockDim.x + threadIdx.x;
    if (n >= N) return;
    float acc0 = 0.f, acc1 = 0.f, acc2 = 0.f, acc3 = 0.f;
    const float* row = x2 + (long long)n * 128;
#pragma unroll 4
    for (int k = 0; k < 128; ++k) {
        float xv = row[k];
        acc0 += xv * W2[k * 4 + 0];
        acc1 += xv * W2[k * 4 + 1];
        acc2 += xv * W2[k * 4 + 2];
        acc3 += xv * W2[k * 4 + 3];
    }
    h2[n * 4 + 0] = acc0; h2[n * 4 + 1] = acc1; h2[n * 4 + 2] = acc2; h2[n * 4 + 3] = acc3;
    as2[n] = acc0 * att_src[0] + acc1 * att_src[1] + acc2 * att_src[2] + acc3 * att_src[3];
    ad2[n] = acc0 * att_dst[0] + acc1 * att_dst[1] + acc2 * att_dst[2] + acc3 * att_dst[3];
}

__global__ void k_edge_max2(const int* __restrict__ ei, int E, long long Etot,
                            const float* __restrict__ as2, const float* __restrict__ ad2,
                            float* __restrict__ ev, unsigned* __restrict__ menc) {
    long long e = (long long)blockIdx.x * blockDim.x + threadIdx.x;
    if (e >= Etot) return;
    int s, d; edge_endpoints(e, ei, E, s, d);
    float lg = lrelu(as2[s] + ad2[d]);
    ev[e] = lg;
    atomicMax(&menc[d], enc_f32(lg));
}

__global__ void k_edge_exp2(const int* __restrict__ ei, int E, long long Etot,
                            const unsigned* __restrict__ menc,
                            float* __restrict__ ev, float* __restrict__ ssum) {
    long long e = (long long)blockIdx.x * blockDim.x + threadIdx.x;
    if (e >= Etot) return;
    int s, d; edge_endpoints(e, ei, E, s, d);
    float ex = __expf(ev[e] - dec_f32(menc[d]));
    ev[e] = ex;
    atomicAdd(&ssum[d], ex);
}

__global__ void k_edge_msg2(const int* __restrict__ ei, int E, long long total,
                            const float* __restrict__ h2, const float* __restrict__ ev,
                            const float* __restrict__ ssum, float* __restrict__ out2) {
    long long idx = (long long)blockIdx.x * blockDim.x + threadIdx.x;
    if (idx >= total) return;
    long long e = idx >> 2;
    int c = (int)(idx & 3);
    int s, d; edge_endpoints(e, ei, E, s, d);
    float alpha = ev[e] / (ssum[d] + 1e-16f);
    atomicAdd(&out2[d * 4 + c], h2[s * 4 + c] * alpha);
}

__global__ void k_softmax_out(const float* __restrict__ out2, const float* __restrict__ b2,
                              float* __restrict__ y, int N) {
    int n = blockIdx.x * blockDim.x + threadIdx.x;
    if (n >= N) return;
    float v[4];
#pragma unroll
    for (int c = 0; c < 4; ++c) v[c] = out2[n * 4 + c] + b2[c];
    float m = fmaxf(fmaxf(v[0], v[1]), fmaxf(v[2], v[3]));
    float s = 0.f;
#pragma unroll
    for (int c = 0; c < 4; ++c) { v[c] = __expf(v[c] - m); s += v[c]; }
    float inv = 1.f / s;
#pragma unroll
    for (int c = 0; c < 4; ++c) y[n * 4 + c] = v[c] * inv;
}

// ---------- launch ----------
static inline unsigned nblk(long long n) { return (unsigned)((n + TPB - 1) / TPB); }

extern "C" void kernel_launch(void* const* d_in, const int* in_sizes, int n_in,
                              void* d_out, int out_size, void* d_ws, size_t ws_size,
                              hipStream_t stream) {
    const float* x        = (const float*)d_in[0];
    const int*   ei       = (const int*)d_in[1];
    const float* W1       = (const float*)d_in[3];
    const float* att_src1 = (const float*)d_in[4];
    const float* att_dst1 = (const float*)d_in[5];
    const float* b1       = (const float*)d_in[6];
    const float* W2       = (const float*)d_in[7];
    const float* att_src2 = (const float*)d_in[8];
    const float* att_dst2 = (const float*)d_in[9];
    const float* b2       = (const float*)d_in[10];

    const int N = in_sizes[0] / 128;
    const int E = in_sizes[1] / 2;
    const long long Etot = (long long)E + N;

    // carve workspace
    size_t off = 0;
    char* ws = (char*)d_ws;
    auto carve = [&](size_t bytes) -> char* {
        char* p = ws + off;
        off += (bytes + 255) & ~(size_t)255;
        return p;
    };
    _Float16* xh   = (_Float16*)carve((size_t)N * 128 * 2);
    _Float16* Wh   = (_Float16*)carve(128 * 128 * 2);
    float*    h1   = (float*)carve((size_t)N * 128 * 4);
    float*    as1  = (float*)carve((size_t)N * 2 * 4);
    float*    ad1  = (float*)carve((size_t)N * 2 * 4);
    unsigned* m1   = (unsigned*)carve((size_t)N * 2 * 4);
    float*    s1   = (float*)carve((size_t)N * 2 * 4);
    float*    ev1  = (float*)carve((size_t)Etot * 2 * 4);
    float*    out1 = (float*)carve((size_t)N * 128 * 4);
    float*    h2   = (float*)carve((size_t)N * 4 * 4);
    float*    as2  = (float*)carve((size_t)N * 4);
    float*    ad2  = (float*)carve((size_t)N * 4);
    unsigned* m2   = (unsigned*)carve((size_t)N * 4);
    float*    s2   = (float*)carve((size_t)N * 4);
    float*    ev2  = (float*)carve((size_t)Etot * 4);
    float*    out2 = (float*)carve((size_t)N * 4 * 4);

    // init
    k_zero_f32<<<nblk((long long)N * 128), TPB, 0, stream>>>(out1, (long long)N * 128);
    k_zero_f32<<<nblk(N * 2), TPB, 0, stream>>>(s1, N * 2);
    k_zero_u32<<<nblk(N * 2), TPB, 0, stream>>>(m1, N * 2);
    k_zero_f32<<<nblk(N), TPB, 0, stream>>>(s2, N);
    k_zero_u32<<<nblk(N), TPB, 0, stream>>>(m2, N);
    k_zero_f32<<<nblk(N * 4), TPB, 0, stream>>>(out2, N * 4);

    // layer 1
    k_cvt_f16<<<nblk((long long)N * 128), TPB, 0, stream>>>(x, xh, (long long)N * 128);
    k_cvt_f16<<<nblk(128 * 128), TPB, 0, stream>>>(W1, Wh, 128 * 128);
    k_gemm1_wmma<<<N / 16, TPB, 0, stream>>>(xh, Wh, h1);
    k_att1<<<nblk(N * 2), TPB, 0, stream>>>(h1, att_src1, att_dst1, as1, ad1, N);
    k_edge_max1<<<nblk(Etot), TPB, 0, stream>>>(ei, E, Etot, as1, ad1, ev1, m1);
    k_edge_exp1<<<nblk(Etot), TPB, 0, stream>>>(ei, E, Etot, m1, ev1, s1);
    k_edge_msg1<<<nblk(Etot * 128), TPB, 0, stream>>>(ei, E, Etot * 128, h1, ev1, s1, out1);
    k_elu_bias<<<nblk((long long)N * 128), TPB, 0, stream>>>(out1, b1, (long long)N * 128);

    // layer 2
    k_layer2_node<<<nblk(N), TPB, 0, stream>>>(out1, W2, att_src2, att_dst2, h2, as2, ad2, N);
    k_edge_max2<<<nblk(Etot), TPB, 0, stream>>>(ei, E, Etot, as2, ad2, ev2, m2);
    k_edge_exp2<<<nblk(Etot), TPB, 0, stream>>>(ei, E, Etot, m2, ev2, s2);
    k_edge_msg2<<<nblk(Etot * 4), TPB, 0, stream>>>(ei, E, Etot * 4, h2, ev2, s2, out2);
    k_softmax_out<<<nblk(N), TPB, 0, stream>>>(out2, b2, (float*)d_out, N);
}
